// JointSentenceBiLSTM12__81114752352625
// MI455X (gfx1250) — matrix operand order repaired and checked by reading, not compiled, not run on hardware
//
#include <hip/hip_runtime.h>
#include <hip/hip_bf16.h>
#include <cstdint>

// ---------------------------------------------------------------------------
// Model dims
// ---------------------------------------------------------------------------
#define Bv 16
#define Lv 128
#define Ev 34
#define Av 36
#define Dv 300
#define DPv 320      // D padded to multiple of 32 for WMMA K
#define Hv 256
#define H2v 512      // 2H
#define G4v 1024     // 4H
#define NPADv 48     // E/A padded to 3 N-tiles of 16

typedef __attribute__((ext_vector_type(16))) _Float16 v16h;
typedef __attribute__((ext_vector_type(8)))  _Float16 v8h;
typedef __attribute__((ext_vector_type(8)))  float    v8f;

#define DEVINL static __device__ __forceinline__

DEVINL float sigmoidf_(float x) { return 1.0f / (1.0f + __expf(-x)); }

// ---------------------------------------------------------------------------
// WMMA fragment helpers (A/B in contiguous k order, wave32)
// ---------------------------------------------------------------------------
// A 16x32 f16: lane&15 = row; lane<16 holds K {k0..k0+15},
// lane>=16 holds K {k0+16..k0+31}.
DEVINL v16h load_a_frag(const _Float16* X, int ld, int row0, int k0) {
  const int lane = threadIdx.x & 31;
  const int row  = row0 + (lane & 15);
  const int kb   = k0 + ((lane & 16) ? 16 : 0);
  const _Float16* p = X + row * ld + kb;
  v8h lo = *(const v8h*)(p);
  v8h hi = *(const v8h*)(p + 8);
  v16h a;
#pragma unroll
  for (int i = 0; i < 8; ++i) { a[i] = lo[i]; a[i + 8] = hi[i]; }
  return a;
}

// B 32x16 f16 with B = W^T (W row-major [N,K]): lane&15 = n, lane bit4 picks
// K half 0..15 vs 16..31; each lane holds 16 contiguous halves of row n of W.
DEVINL v16h load_b_frag(const _Float16* W, int ld, int nrow0, int k0) {
  const int lane = threadIdx.x & 31;
  const int n    = nrow0 + (lane & 15);
  const int kb   = k0 + ((lane & 16) ? 16 : 0);
  return *(const v16h*)(W + n * ld + kb);
}

DEVINL v8f wmma16(v16h a, v16h b, v8f c) {
  // (neg_a, A, neg_b, B, c_mod, C, reuse_a, reuse_b)
  return __builtin_amdgcn_wmma_f32_16x16x32_f16(false, a, false, b, (short)0, c,
                                                false, false);
}

// C/D 16x16 f32: VGPR r, lane<16 -> row=r, lane>=16 -> row=8+r, col=lane&15
DEVINL void store_acc(float* C, int ld, int row0, int col0, v8f acc) {
  const int lane = threadIdx.x & 31;
  const int r0 = row0 + ((lane & 16) ? 8 : 0);
  const int c  = col0 + (lane & 15);
#pragma unroll
  for (int r = 0; r < 8; ++r) C[(r0 + r) * ld + c] = acc[r];
}

// ---------------------------------------------------------------------------
// Kernel 0: fp32 -> fp16 weight repack with zero padding
// ---------------------------------------------------------------------------
__global__ void convert_f16_kernel(const float* __restrict__ src,
                                   _Float16* __restrict__ dst,
                                   int srcRows, int srcCols, int srcLd,
                                   int dstRows, int dstCols) {
  int idx = blockIdx.x * blockDim.x + threadIdx.x;
  int total = dstRows * dstCols;
  if (idx >= total) return;
  int c = idx % dstCols;
  int r = idx / dstCols;
  float v = 0.0f;
  if (r < srcRows && c < srcCols) v = src[r * srcLd + c];
  dst[idx] = (_Float16)v;
}

// ---------------------------------------------------------------------------
// Kernel 1: embedding gather -> Xh [L*B, DP] fp16 (row = l*B + b), zero pad
// ---------------------------------------------------------------------------
__global__ void embed_kernel(const int* __restrict__ ids,
                             const float* __restrict__ emb,
                             _Float16* __restrict__ Xh) {
  int idx = blockIdx.x * blockDim.x + threadIdx.x;
  if (idx >= Lv * Bv * DPv) return;
  int k   = idx % DPv;
  int row = idx / DPv;          // l*B + b
  int l = row >> 4;
  int b = row & 15;
  float v = 0.0f;
  if (k < Dv) v = emb[(size_t)ids[b * Lv + l] * Dv + k];
  Xh[idx] = (_Float16)v;
}

// ---------------------------------------------------------------------------
// Kernel 2: input projection, both directions, stored TRANSPOSED:
//   P[pos][col][b]  (col = gate*256+u, b = batch)  so that the LSTM's
// accumulator preload is 8 contiguous floats per lane (v8f, 2x b128).
// M = L*B = 2048 (128 M-tiles == one pos each), N = 1024 (64 tiles), K = 320.
// ---------------------------------------------------------------------------
__global__ void gemm_p_kernel(const _Float16* __restrict__ Xh,
                              const _Float16* __restrict__ Wih16_f,
                              const _Float16* __restrict__ Wih16_b,
                              const float* __restrict__ bih_f,
                              const float* __restrict__ bhh_f,
                              const float* __restrict__ bih_b,
                              const float* __restrict__ bhh_b,
                              float* __restrict__ Pf,
                              float* __restrict__ Pb) {
  int wave  = (blockIdx.x * blockDim.x + threadIdx.x) >> 5;
  int ntile = wave & 63;
  int mtile = (wave >> 6) & 127;   // == sequence position pos
  int dir   = wave >> 13;
  const _Float16* W = dir ? Wih16_b : Wih16_f;
  const float* b1 = dir ? bih_b : bih_f;
  const float* b2 = dir ? bhh_b : bhh_f;
  float* P = dir ? Pb : Pf;

  v8f acc = {};
#pragma unroll
  for (int kt = 0; kt < DPv / 32; ++kt) {
    v16h a = load_a_frag(Xh, DPv, mtile * 16, kt * 32);
    v16h b = load_b_frag(W, DPv, ntile * 16, kt * 32);
    acc = wmma16(a, b, acc);
  }
  const int lane = threadIdx.x & 31;
  const int col  = ntile * 16 + (lane & 15);
  const int r0   = (lane & 16) ? 8 : 0;      // batch sub-block
  float bias = b1[col] + b2[col];
#pragma unroll
  for (int r = 0; r < 8; ++r) acc[r] += bias;
  // transposed store: 8 contiguous floats (batches r0..r0+7) per lane
  *(v8f*)(P + ((size_t)mtile * G4v + col) * Bv + r0) = acc;
}

// ---------------------------------------------------------------------------
// Kernel 3: bidirectional LSTM recurrence. grid = 2 (dir), block = 512 (16
// waves). Wave w owns hidden-unit columns 16w..16w+15 for ALL FOUR gates
// (N-tiles w, 16+w, 32+w, 48+w) so the cell update is register-resident.
// h lives in 8KB LDS (fp16); c lives in registers. Whh fp16 streams from L2.
// P is in transposed [pos][col][b] layout -> v8f coalesced preloads.
// ---------------------------------------------------------------------------
__global__ void lstm_kernel(const float* __restrict__ Pf,
                            const float* __restrict__ Pb,
                            const _Float16* __restrict__ Whh16_f,
                            const _Float16* __restrict__ Whh16_b,
                            _Float16* __restrict__ Hs16) {
  __shared__ __align__(32) _Float16 hsh[Bv * Hv];   // [16][256]
  const int dir  = blockIdx.x;
  const float* P = dir ? Pb : Pf;
  const _Float16* Whh = dir ? Whh16_b : Whh16_f;
  const int tid  = threadIdx.x;
  const int wave = tid >> 5;        // 0..15
  const int lane = tid & 31;
  const int r0   = (lane & 16) ? 8 : 0;
  const int u    = wave * 16 + (lane & 15);   // hidden unit this lane owns

  for (int i = tid; i < Bv * Hv; i += blockDim.x) hsh[i] = (_Float16)0.0f;
  v8f c = {};
  __syncthreads();

  for (int s = 0; s < Lv; ++s) {
    int pos = dir ? (Lv - 1 - s) : s;
    const float* Pt = P + (size_t)pos * G4v * Bv;

    // accumulators preloaded with input projection (z = P + h @ Whh^T)
    v8f zi = *(const v8f*)(Pt + (0 * Hv + u) * Bv + r0);
    v8f zf = *(const v8f*)(Pt + (1 * Hv + u) * Bv + r0);
    v8f zg = *(const v8f*)(Pt + (2 * Hv + u) * Bv + r0);
    v8f zo = *(const v8f*)(Pt + (3 * Hv + u) * Bv + r0);

    if (s + 1 < Lv) {  // warm next step's 64KB P slice into cache
      int npos = dir ? (Lv - 2 - s) : (s + 1);
      __builtin_prefetch(P + (size_t)npos * G4v * Bv + tid * 32, 0, 1);
    }

#pragma unroll
    for (int kt = 0; kt < Hv / 32; ++kt) {
      v16h a  = load_a_frag(hsh, Hv, 0, kt * 32);
      v16h bi = load_b_frag(Whh, Hv, 0 * Hv + wave * 16, kt * 32);
      v16h bf = load_b_frag(Whh, Hv, 1 * Hv + wave * 16, kt * 32);
      v16h bg = load_b_frag(Whh, Hv, 2 * Hv + wave * 16, kt * 32);
      v16h bo = load_b_frag(Whh, Hv, 3 * Hv + wave * 16, kt * 32);
      zi = wmma16(a, bi, zi);
      zf = wmma16(a, bf, zf);
      zg = wmma16(a, bg, zg);
      zo = wmma16(a, bo, zo);
    }
    __syncthreads();   // all waves finished reading hsh for this step

#pragma unroll
    for (int r = 0; r < 8; ++r) {
      float ii = sigmoidf_(zi[r]);
      float ff = sigmoidf_(zf[r]);
      float gg = tanhf(zg[r]);
      float oo = sigmoidf_(zo[r]);
      float cv = ff * c[r] + ii * gg;
      c[r] = cv;
      float hv = oo * tanhf(cv);
      _Float16 h16 = (_Float16)hv;
      int b = r0 + r;
      hsh[b * Hv + u] = h16;
      Hs16[((size_t)pos * Bv + b) * H2v + dir * Hv + u] = h16;
    }
    __syncthreads();   // h visible before next step's A loads
  }
}

// ---------------------------------------------------------------------------
// Kernel 4: hoisted scan bases.
//   Hb [j*B+b, 48] = hidden . Wa[:, 0:512]
//   Tb [i*B+b, 48] = hidden . Wa[:, 512:1024]
//   Eb [i*B+b, 48] = hidden . We[:, 0:512]
// M = 2048 (128 tiles), N = 48 (3 tiles), K = 512; 3*128*3 = 1152 waves.
// ---------------------------------------------------------------------------
__global__ void gemm_base_kernel(const _Float16* __restrict__ Hs16,
                                 const _Float16* __restrict__ Wa16,
                                 const _Float16* __restrict__ We16,
                                 float* __restrict__ Hb,
                                 float* __restrict__ Tb,
                                 float* __restrict__ Eb) {
  int wave  = (blockIdx.x * blockDim.x + threadIdx.x) >> 5;
  int ntile = wave % 3;
  int mtile = (wave / 3) % 128;
  int mat   = wave / 384;
  const _Float16* W; int ld, koff; float* Out;
  if (mat == 0)      { W = Wa16; ld = 1024; koff = 0;   Out = Hb; }
  else if (mat == 1) { W = Wa16; ld = 1024; koff = 512; Out = Tb; }
  else               { W = We16; ld = 512;  koff = 0;   Out = Eb; }

  v8f acc = {};
#pragma unroll
  for (int kt = 0; kt < H2v / 32; ++kt) {
    v16h a = load_a_frag(Hs16, H2v, mtile * 16, kt * 32);
    v16h b = load_b_frag(W, ld, ntile * 16, koff + kt * 32);
    acc = wmma16(a, b, acc);
  }
  store_acc(Out, NPADv, mtile * 16, ntile * 16, acc);
}

// ---------------------------------------------------------------------------
// Kernel 5: sequential prediction scan. One block per batch b (16 blocks,
// 128 threads = one per position j). Gate one-hot contributions to the
// logits are maintained incrementally (add a column of Wa/We when a gate
// bit flips 0 -> 1), so per-step work is O(L*A) instead of O(L*1092*A).
// ---------------------------------------------------------------------------
__global__ void scan_kernel(const float* __restrict__ Hb,
                            const float* __restrict__ Tb,
                            const float* __restrict__ Eb,
                            const float* __restrict__ We,   // [34, 545]
                            const float* __restrict__ be,   // [34]
                            const float* __restrict__ Wa,   // [36, 1092]
                            const float* __restrict__ ba,   // [36]
                            float* __restrict__ out_ev,     // [B,L,E]
                            float* __restrict__ out_arg) {  // [B,L,L,A]
  __shared__ float contrib[Lv][Av];            // per-position gate contribution
  __shared__ unsigned long long gta[Lv];       // g_trg_arg bit mask (E-1 bits)
  __shared__ unsigned long long garg[Lv];      // g_arg bit mask     (A-1 bits)
  __shared__ float evx[Ev];                    // g_trg contribution to ev logit
  __shared__ float evlog[Ev];
  __shared__ unsigned long long gtrg;
  __shared__ int evpred, newTrg;
  __shared__ int argpred[Lv];

  const int b   = blockIdx.x;
  const int tid = threadIdx.x;                 // = position j, 0..127

  for (int a = 0; a < Av; ++a) contrib[tid][a] = 0.0f;
  gta[tid]  = 0ull;
  garg[tid] = 0ull;
  if (tid < Ev) evx[tid] = 0.0f;
  if (tid == 0) gtrg = 0ull;
  __syncthreads();

  for (int i = 0; i < Lv; ++i) {
    // ---- event logits ----
    if (tid < Ev) {
      float v = Eb[(i * Bv + b) * NPADv + tid] + evx[tid] + be[tid];
      evlog[tid] = v;
      out_ev[((size_t)b * Lv + i) * Ev + tid] = v;
    }
    __syncthreads();
    if (tid == 0) {
      float best = evlog[0]; int bi = 0;
      for (int e = 1; e < Ev; ++e)
        if (evlog[e] > best) { best = evlog[e]; bi = e; }
      evpred = bi;
      newTrg = 0;
      if (bi > 0) {
        unsigned long long bit = 1ull << (bi - 1);
        if (!(gtrg & bit)) { gtrg |= bit; newTrg = 1; }
      }
    }
    __syncthreads();
    const int ev = evpred;

    // ---- argument logits for position j = tid ----
    {
      const float* hb = Hb + (tid * Bv + b) * NPADv;
      const float* tb = Tb + (i * Bv + b) * NPADv;
      float* op = out_arg + (((size_t)(b * Lv + i)) * Lv + tid) * Av;
      float best = 0.0f; int bi = 0;
      for (int a = 0; a < Av; ++a) {
        float v = hb[a] + tb[a] + contrib[tid][a] + ba[a];
        op[a] = v;
        if (a == 0) { best = v; bi = 0; }
        else if (v > best) { best = v; bi = a; }
      }
      argpred[tid] = bi;
    }

    // ---- gate updates (take effect next step) ----
    if (ev > 0) {
      int k = ev - 1;
      int ap = argpred[tid];
      if (ap > 0) {
        unsigned long long bit = 1ull << k;
        if (!(gta[tid] & bit)) {
          gta[tid] |= bit;
          for (int a = 0; a < Av; ++a)
            contrib[tid][a] += Wa[a * 1092 + 1059 + k];      // g_trg_arg cols
        }
        unsigned long long bit2 = 1ull << (ap - 1);
        if (!(garg[tid] & bit2)) {
          garg[tid] |= bit2;
          for (int a = 0; a < Av; ++a)
            contrib[tid][a] += Wa[a * 1092 + 1024 + (ap - 1)]; // g_arg cols
        }
      }
      if (newTrg && tid < Ev) evx[tid] += We[tid * 545 + 512 + k];
    }
    __syncthreads();
  }
}

// ---------------------------------------------------------------------------
// Host launcher
// ---------------------------------------------------------------------------
extern "C" void kernel_launch(void* const* d_in, const int* in_sizes, int n_in,
                              void* d_out, int out_size, void* d_ws, size_t ws_size,
                              hipStream_t stream) {
  const int*   ids   = (const int*)  d_in[0];
  const float* emb   = (const float*)d_in[1];
  const float* Wih_f = (const float*)d_in[2];
  const float* Whh_f = (const float*)d_in[3];
  const float* bih_f = (const float*)d_in[4];
  const float* bhh_f = (const float*)d_in[5];
  const float* Wih_b = (const float*)d_in[6];
  const float* Whh_b = (const float*)d_in[7];
  const float* bih_b = (const float*)d_in[8];
  const float* bhh_b = (const float*)d_in[9];
  const float* We    = (const float*)d_in[10];
  const float* be    = (const float*)d_in[11];
  const float* Wa    = (const float*)d_in[12];
  const float* ba    = (const float*)d_in[13];

  char* ws = (char*)d_ws;
  size_t off = 0;
  auto alloc = [&](size_t bytes) { char* p = ws + off; off += (bytes + 255) & ~(size_t)255; return p; };

  _Float16* Xh      = (_Float16*)alloc((size_t)Lv * Bv * DPv * 2);
  _Float16* Wih16_f = (_Float16*)alloc((size_t)G4v * DPv * 2);
  _Float16* Wih16_b = (_Float16*)alloc((size_t)G4v * DPv * 2);
  _Float16* Whh16_f = (_Float16*)alloc((size_t)G4v * Hv * 2);
  _Float16* Whh16_b = (_Float16*)alloc((size_t)G4v * Hv * 2);
  _Float16* Wa16    = (_Float16*)alloc((size_t)NPADv * 1024 * 2);
  _Float16* We16    = (_Float16*)alloc((size_t)NPADv * 512 * 2);
  float*    Pf      = (float*)   alloc((size_t)Lv * Bv * G4v * 4);
  float*    Pb      = (float*)   alloc((size_t)Lv * Bv * G4v * 4);
  _Float16* Hs16    = (_Float16*)alloc((size_t)Lv * Bv * H2v * 2);
  float*    Hbase   = (float*)   alloc((size_t)Lv * Bv * NPADv * 4);
  float*    Tbase   = (float*)   alloc((size_t)Lv * Bv * NPADv * 4);
  float*    Ebase   = (float*)   alloc((size_t)Lv * Bv * NPADv * 4);

  float* out_ev  = (float*)d_out;
  float* out_arg = out_ev + (size_t)Bv * Lv * Ev;

  const int TB = 256;
  auto grid = [](int total, int tb) { return (total + tb - 1) / tb; };

  // 0) weight repack to fp16 (padded)
  convert_f16_kernel<<<grid(G4v * DPv, TB), TB, 0, stream>>>(Wih_f, Wih16_f, G4v, Dv, Dv, G4v, DPv);
  convert_f16_kernel<<<grid(G4v * DPv, TB), TB, 0, stream>>>(Wih_b, Wih16_b, G4v, Dv, Dv, G4v, DPv);
  convert_f16_kernel<<<grid(G4v * Hv, TB), TB, 0, stream>>>(Whh_f, Whh16_f, G4v, Hv, Hv, G4v, Hv);
  convert_f16_kernel<<<grid(G4v * Hv, TB), TB, 0, stream>>>(Whh_b, Whh16_b, G4v, Hv, Hv, G4v, Hv);
  convert_f16_kernel<<<grid(NPADv * 1024, TB), TB, 0, stream>>>(Wa, Wa16, Av, 1024, 1092, NPADv, 1024);
  convert_f16_kernel<<<grid(NPADv * 512, TB), TB, 0, stream>>>(We, We16, Ev, 512, 545, NPADv, 512);

  // 1) embedding gather (fp16, padded K)
  embed_kernel<<<grid(Lv * Bv * DPv, TB), TB, 0, stream>>>(ids, emb, Xh);

  // 2) input projections, both directions: 16384 WMMA tiles (transposed out)
  gemm_p_kernel<<<(2 * 128 * 64 * 32) / TB, TB, 0, stream>>>(
      Xh, Wih16_f, Wih16_b, bih_f, bhh_f, bih_b, bhh_b, Pf, Pb);

  // 3) bidirectional LSTM recurrence (sequential scan, 1 WGP per direction)
  lstm_kernel<<<2, 512, 0, stream>>>(Pf, Pb, Whh16_f, Whh16_b, Hs16);

  // 4) hoisted logit bases: 1152 WMMA tiles
  gemm_base_kernel<<<(3 * 128 * 3 * 32) / TB, TB, 0, stream>>>(
      Hs16, Wa16, We16, Hbase, Tbase, Ebase);

  // 5) prediction scan with incremental gate contributions
  scan_kernel<<<Bv, Lv, 0, stream>>>(Hbase, Tbase, Ebase, We, be, Wa, ba,
                                     out_ev, out_arg);
}
